// AdjAttenAggr_9388798509098
// MI455X (gfx1250) — matrix operand order repaired and checked by reading, not compiled, not run on hardware
//
#include <hip/hip_runtime.h>
#include <hip/hip_bf16.h>

typedef __attribute__((ext_vector_type(16))) _Float16 v16h;
typedef __attribute__((ext_vector_type(8)))  float    v8f;

#define NROWS 8192       // N
#define NKEYS 8192       // M
#define QDIM  512
#define KDIM  512
#define MID   256
#define CHUNK 128        // keys per flash chunk
#define INV_SCALE (1.0f/16.0f)   // 1/sqrt(MID)
#define NEGV  (-4294967296.0f)   // -float(1<<32)

// ---------------------------------------------------------------------------
// A-matrix fragment layout (16-bit A, 16x32, contiguous k order used here):
//   lane L: row = L&15, half = L>>4
//   a[j]   (j=0..7)  -> k = kbase + 16*half + j
//   a[8+j] (j=0..7)  -> k = kbase + 16*half + 8 + j
// B-matrix fragment layout (16-bit B, 32x16):
//   lane L: col = L&15, half = L>>4
//   b[j]   (j=0..15) -> k = kbase + 16*half + j
// C/D (f32 16x16): VGPR r, lane L: row = r + 8*(L>>4), col = L&15
// ---------------------------------------------------------------------------

// ============================================================================
// Kernel 1: fused Q/K projection GEMMs. One wave per 16x16 output tile.
//   tiles 0..8191      -> Q = main_feat @ Wq^T + bq   (f16 out)
//   tiles 8192..16383  -> K = other_feat @ Wk^T + bk  (f16 out)
// ============================================================================
__global__ __launch_bounds__(256) void proj_kernel(
    const float* __restrict__ main_feat, const float* __restrict__ other_feat,
    const float* __restrict__ Wq, const float* __restrict__ bq,
    const float* __restrict__ Wk, const float* __restrict__ bk,
    _Float16* __restrict__ Qh, _Float16* __restrict__ Kh)
{
    const int wave = threadIdx.x >> 5;
    const int lane = threadIdx.x & 31;
    const int ln   = lane & 15;
    const int half = lane >> 4;

    int tile = blockIdx.x * 8 + wave;
    const bool isK = tile >= 8192;
    const int t = isK ? tile - 8192 : tile;
    const int m0 = (t >> 4) * 16;       // 16 column-tiles across MID=256
    const int n0 = (t & 15) * 16;

    const float* __restrict__ X    = isK ? other_feat : main_feat;
    const float* __restrict__ W    = isK ? Wk : Wq;
    const float* __restrict__ bias = isK ? bk : bq;
    _Float16*    __restrict__ Y    = isK ? Kh : Qh;

    const float* xrow = X + (long)(m0 + ln) * QDIM;   // A: lane = row
    const float* wrow = W + (long)(n0 + ln) * QDIM;   // B: lane = out column

    v8f c = {};
    #pragma unroll 4
    for (int kk = 0; kk < QDIM; kk += 32) {
        v16h a, b;
        const float* ap = xrow + kk + 16 * half;
        #pragma unroll
        for (int j = 0; j < 8; ++j) {
            a[j]     = (_Float16)ap[j];
            a[8 + j] = (_Float16)ap[8 + j];
        }
        const float* bp = wrow + kk + 16 * half;
        #pragma unroll
        for (int j = 0; j < 16; ++j) b[j] = (_Float16)bp[j];
        c = __builtin_amdgcn_wmma_f32_16x16x32_f16(false, a, false, b,
                                                   (short)0, c, false, false);
    }

    const float bv = bias[n0 + ln];
    #pragma unroll
    for (int r = 0; r < 8; ++r) {
        const int row = r + 8 * half;
        Y[(long)(m0 + row) * MID + n0 + ln] = (_Float16)(c[r] + bv);
    }
}

// ============================================================================
// Kernel 2: V^T = (fix ⊙ other)^T in f16, via LDS-tiled transpose so the
// flash kernel's B-fragments are contiguous per-lane reads.
// ============================================================================
__global__ __launch_bounds__(256) void vt_kernel(
    const float* __restrict__ other, const float* __restrict__ fix,
    _Float16* __restrict__ VT)
{
    __shared__ float tile[32][33];
    const int t  = threadIdx.x;
    const int bm = (blockIdx.x & 255) * 32;   // 8192/32 = 256 m-tiles
    const int bc = (blockIdx.x >> 8) * 32;    // 512/32  = 16  c-tiles

    #pragma unroll
    for (int i = 0; i < 4; ++i) {
        const int e  = t + 256 * i;
        const int mr = e >> 5, cc = e & 31;
        tile[mr][cc] = other[(long)(bm + mr) * KDIM + bc + cc] * fix[bm + mr];
    }
    __syncthreads();
    #pragma unroll
    for (int i = 0; i < 4; ++i) {
        const int e  = t + 256 * i;
        const int cr = e >> 5, mm = e & 31;
        VT[(long)(bc + cr) * NKEYS + bm + mm] = (_Float16)tile[mm][cr];
    }
}

// ============================================================================
// Kernel 3: flash attention. Block = 8 waves = 16 query rows.
//   per 128-key chunk: each wave -> one 16x16 S subtile (8 WMMAs over MID),
//   mask+scale into LDS, cooperative rowmax / exp / rowsum, online-softmax
//   rescale, then each wave accumulates its 64 output columns (16 WMMAs).
// ============================================================================
__global__ __launch_bounds__(256) void flash_kernel(
    const _Float16* __restrict__ Qh, const _Float16* __restrict__ Kh,
    const _Float16* __restrict__ VT, const unsigned char* __restrict__ mask,
    float* __restrict__ out)
{
    __shared__ float    Sl[16 * CHUNK];
    __shared__ _Float16 Pl[16 * CHUNK];
    __shared__ float    red[16 * 16];
    __shared__ float    m_run[16], l_run[16], m_new[16], fS[16];

    const int tid  = threadIdx.x;
    const int wave = tid >> 5;
    const int lane = tid & 31;
    const int ln   = lane & 15;
    const int half = lane >> 4;
    const long qbase = (long)blockIdx.x * 16;

    if (tid < 16) { m_run[tid] = -__builtin_inff(); l_run[tid] = 0.0f; }

    // Whole 16xMID Q tile lives in registers as 8 A-fragments (64 VGPRs/lane).
    v16h Aq[8];
    {
        const _Float16* qrow = Qh + (qbase + ln) * MID;
        #pragma unroll
        for (int kf = 0; kf < 8; ++kf) {
            const _Float16* ap = qrow + kf * 32 + 16 * half;
            #pragma unroll
            for (int j = 0; j < 8; ++j) {
                Aq[kf][j]     = ap[j];
                Aq[kf][8 + j] = ap[8 + j];
            }
        }
    }

    v8f acc[4] = {};   // 16 rows x 64 cols per wave
    __syncthreads();

    for (int key0 = 0; key0 < NKEYS; key0 += CHUNK) {
        // --- S subtile: rows = block's 16 q-rows, cols = wave's 16 keys ----
        v8f s = {};
        {
            const _Float16* krow = Kh + (long)(key0 + wave * 16 + ln) * MID;
            #pragma unroll
            for (int kf = 0; kf < 8; ++kf) {
                v16h b;
                const _Float16* bp = krow + kf * 32 + 16 * half;
                #pragma unroll
                for (int j = 0; j < 16; ++j) b[j] = bp[j];
                s = __builtin_amdgcn_wmma_f32_16x16x32_f16(false, Aq[kf], false, b,
                                                           (short)0, s, false, false);
            }
        }
        // --- scale + mask -> LDS -------------------------------------------
        {
            const int col = wave * 16 + ln;
            const unsigned char* mp = mask + qbase * NKEYS + key0 + col;
            #pragma unroll
            for (int r = 0; r < 8; ++r) {
                const int row = r + 8 * half;
                const float v = s[r] * INV_SCALE;
                const unsigned char mb = mp[(long)row * NKEYS];
                Sl[row * CHUNK + col] = mb ? NEGV : v;
            }
        }
        __syncthreads();
        // --- row max (2-level reduction) -----------------------------------
        {
            const int row = tid & 15, seg = tid >> 4;
            float mx = -__builtin_inff();
            #pragma unroll
            for (int j = 0; j < 8; ++j)
                mx = fmaxf(mx, Sl[row * CHUNK + seg * 8 + j]);
            red[row * 16 + seg] = mx;
        }
        __syncthreads();
        if (tid < 16) {
            float mx = -__builtin_inff();
            #pragma unroll
            for (int j = 0; j < 16; ++j) mx = fmaxf(mx, red[tid * 16 + j]);
            const float mn = fmaxf(m_run[tid], mx);
            fS[tid]    = __expf(m_run[tid] - mn);   // exp(-inf)=0 on 1st chunk
            m_new[tid] = mn;
            m_run[tid] = mn;
        }
        __syncthreads();
        // --- P = exp(S - m_new), f16 to LDS, partial row sums --------------
        {
            const int row = tid & 15, seg = tid >> 4;
            const float mn = m_new[row];
            float sum = 0.0f;
            #pragma unroll
            for (int j = 0; j < 8; ++j) {
                const int c = seg * 8 + j;
                const float p = __expf(Sl[row * CHUNK + c] - mn);
                Pl[row * CHUNK + c] = (_Float16)p;
                sum += p;
            }
            red[row * 16 + seg] = sum;
        }
        __syncthreads();
        if (tid < 16) {
            float sm = 0.0f;
            #pragma unroll
            for (int j = 0; j < 16; ++j) sm += red[tid * 16 + j];
            l_run[tid] = l_run[tid] * fS[tid] + sm;
        }
        // --- online rescale of accumulators --------------------------------
        float fr[8];
        #pragma unroll
        for (int r = 0; r < 8; ++r) fr[r] = fS[r + 8 * half];
        #pragma unroll
        for (int ct = 0; ct < 4; ++ct)
            #pragma unroll
            for (int r = 0; r < 8; ++r) acc[ct][r] *= fr[r];
        // --- O += P @ V : wave owns output cols [64*wave, 64*wave+64) ------
        const int n0 = wave * 64;
        #pragma unroll
        for (int ks = 0; ks < 4; ++ks) {
            v16h a;   // A-frag of P from LDS (lane = q-row)
            const _Float16* pp = &Pl[ln * CHUNK + ks * 32];
            #pragma unroll
            for (int j = 0; j < 8; ++j) {
                a[j]     = pp[16 * half + j];
                a[8 + j] = pp[16 * half + 8 + j];
            }
            #pragma unroll
            for (int ct = 0; ct < 4; ++ct) {
                const _Float16* vp = VT + (long)(n0 + ct * 16 + ln) * NKEYS
                                        + key0 + ks * 32 + 16 * half;
                v16h b;
                #pragma unroll
                for (int j = 0; j < 16; ++j) b[j] = vp[j];
                acc[ct] = __builtin_amdgcn_wmma_f32_16x16x32_f16(false, a, false, b,
                                                                 (short)0, acc[ct],
                                                                 false, false);
            }
        }
        __syncthreads();   // protect Sl/Pl before next chunk overwrites
    }

    // --- epilogue: divide by softmax denominator, store f32 ----------------
    float invl[8];
    #pragma unroll
    for (int r = 0; r < 8; ++r) invl[r] = 1.0f / l_run[r + 8 * half];
    #pragma unroll
    for (int ct = 0; ct < 4; ++ct) {
        const int col = wave * 64 + ct * 16 + ln;
        #pragma unroll
        for (int r = 0; r < 8; ++r) {
            const int row = r + 8 * half;
            out[(qbase + row) * KDIM + col] = acc[ct][r] * invl[r];
        }
    }
}

// ============================================================================
extern "C" void kernel_launch(void* const* d_in, const int* in_sizes, int n_in,
                              void* d_out, int out_size, void* d_ws, size_t ws_size,
                              hipStream_t stream) {
    const float*         main_feat  = (const float*)d_in[0];
    const float*         other_feat = (const float*)d_in[1];
    const float*         fix_feat   = (const float*)d_in[2];
    const unsigned char* mask       = (const unsigned char*)d_in[3]; // bool
    const float*         Wq         = (const float*)d_in[4];
    const float*         bq         = (const float*)d_in[5];
    const float*         Wk         = (const float*)d_in[6];
    const float*         bk         = (const float*)d_in[7];
    float*               out        = (float*)d_out;

    char* ws = (char*)d_ws;
    _Float16* Qh = (_Float16*)ws;                                   // 4 MB
    _Float16* Kh = (_Float16*)(ws + (size_t)NROWS * MID * 2);       // 4 MB
    _Float16* VT = (_Float16*)(ws + (size_t)NROWS * MID * 4);       // 8 MB

    // 16384 16x16 tiles (Q + K), 8 waves per block
    proj_kernel<<<2048, 256, 0, stream>>>(main_feat, other_feat,
                                          Wq, bq, Wk, bk, Qh, Kh);
    // 256 x 16 transpose tiles
    vt_kernel<<<4096, 256, 0, stream>>>(other_feat, fix_feat, VT);
    // one block per 16 query rows
    flash_kernel<<<512, 256, 0, stream>>>(Qh, Kh, VT, mask, out);
}
